// QuantMultiHeadSelfAttention_28527172780143
// MI455X (gfx1250) — compile-verified, matrix operand order checked by reading, not run on hardware
//
#include <hip/hip_runtime.h>

typedef __attribute__((ext_vector_type(16))) __bf16 v16bf;
typedef __attribute__((ext_vector_type(8)))  float  v8f;
typedef __attribute__((ext_vector_type(4)))  unsigned int uint4v;
typedef __attribute__((ext_vector_type(4)))  int i4v;
typedef unsigned short u16;

static constexpr int BM = 128;   // workgroup M tile
static constexpr int BN = 128;   // workgroup N tile
static constexpr int BK = 32;    // K step (wmma 16x16x32 bf16)
static constexpr int LDST = 40;  // padded LDS row stride (bf16): 80B/row -> 16B-aligned, conflict-free (20*r mod 64 distinct)
static constexpr int NBUF = 3;   // async pipeline depth (3 x (A+B) tiles = 60KB LDS)

#if __has_builtin(__builtin_amdgcn_global_load_async_to_lds_b128)
#define ASYNC_LDS 1
#endif

__device__ __forceinline__ u16 f2bf(float f) {
    unsigned u = __float_as_uint(f);
    u += 0x7FFFu + ((u >> 16) & 1u);   // round-to-nearest-even
    return (u16)(u >> 16);
}
__device__ __forceinline__ float bf2f(u16 h) {
    return __uint_as_float(((unsigned)h) << 16);
}

// One 16-byte chunk: global -> LDS. Async (ASYNCcnt-tracked, no VGPR staging) when available.
__device__ __forceinline__ void copy16(const u16* __restrict__ g, u16* __restrict__ l) {
#ifdef ASYNC_LDS
    __builtin_amdgcn_global_load_async_to_lds_b128(
        (i4v*)g,
        (__attribute__((address_space(3))) i4v*)l, 0, 0);
#else
    *reinterpret_cast<uint4v*>(l) = *reinterpret_cast<const uint4v*>(g);
#endif
}

template<int N> __device__ __forceinline__ void waitAsync() {
#ifdef ASYNC_LDS
#if __has_builtin(__builtin_amdgcn_s_wait_asynccnt)
    __builtin_amdgcn_s_wait_asynccnt(N);
#else
    asm volatile("s_wait_asynccnt %0" :: "n"(N));
#endif
#endif
}

// Load one 16x32 bf16 fragment (A layout; B 32x16 mirrors it with N in place of M)
// lane L: row = L%16, half = L/16; VGPR 0-3: K in [half*8, half*8+8); VGPR 4-7: +16.
__device__ __forceinline__ v16bf ldsFrag(const u16* tile, int row0, int lane) {
    const int r  = row0 + (lane & 15);
    const int h8 = (lane >> 4) * 8;
    union { uint4v u[2]; v16bf b; } u;
    u.u[0] = *reinterpret_cast<const uint4v*>(tile + r * LDST + h8);
    u.u[1] = *reinterpret_cast<const uint4v*>(tile + r * LDST + 16 + h8);
    return u.b;
}

// C = A(MxK) * Bt(NxK)^T [+ bias], bf16 in, f32 accum, bf16 or f32 out.
// Batched via grid.z: z -> outer = z/innerCount, inner = z%innerCount.
template<bool OUT_BF16, bool TRANS_OUT, bool HAS_BIAS>
__global__ __launch_bounds__(256)
void gemm_bf16_wmma(const u16* __restrict__ A, const u16* __restrict__ Bt,
                    void* __restrict__ Cout, const float* __restrict__ bias,
                    int M, int N, int K, int lda, int ldb, int ldc,
                    int innerCount,
                    long long aOuter, long long aInner,
                    long long bOuter, long long bInner,
                    long long cOuter, long long cInner,
                    int biasInner)
{
    __shared__ u16 lA[NBUF][BM * LDST];
    __shared__ u16 lB[NBUF][BN * LDST];

    const int z     = blockIdx.z;
    const int outer = z / innerCount;
    const int inner = z - outer * innerCount;
    A  += outer * aOuter + inner * aInner;
    Bt += outer * bOuter + inner * bInner;
    u16*   Cb = reinterpret_cast<u16*>(Cout)   + outer * cOuter + inner * cInner;
    float* Cf = reinterpret_cast<float*>(Cout) + outer * cOuter + inner * cInner;
    const float* bptr = HAS_BIAS ? (bias + (long long)inner * biasInner) : nullptr;

    const int tid   = threadIdx.x;
    const int lane  = tid & 31;
    const int wave  = tid >> 5;        // 8 waves
    const int waveM = (wave & 3) * 32; // 4 x 2 wave grid, 32x64 per wave
    const int waveN = (wave >> 2) * 64;

    const int m0 = blockIdx.y * BM;
    const int n0 = blockIdx.x * BN;

    // cooperative loader: 2 x 16B chunks per thread per (A,B) tile -> 4 async instrs/thread/tile
    const int c0 = tid, c1 = tid + 256;
    const int ar0 = c0 >> 2, ak0 = (c0 & 3) * 8;
    const int ar1 = c1 >> 2, ak1 = (c1 & 3) * 8;

    const u16* gA0 = A  + (long long)(m0 + ar0) * lda + ak0;
    const u16* gA1 = A  + (long long)(m0 + ar1) * lda + ak1;
    const u16* gB0 = Bt + (long long)(n0 + ar0) * ldb + ak0;
    const u16* gB1 = Bt + (long long)(n0 + ar1) * ldb + ak1;

    const int nk = K / BK;

    auto issueTile = [&](int kt, int buf) {
        const int k0 = kt * BK;
        copy16(gA0 + k0, &lA[buf][ar0 * LDST + ak0]);
        copy16(gA1 + k0, &lA[buf][ar1 * LDST + ak1]);
        copy16(gB0 + k0, &lB[buf][ar0 * LDST + ak0]);
        copy16(gB1 + k0, &lB[buf][ar1 * LDST + ak1]);
    };

    // prologue: tiles 0 and 1 in flight
    issueTile(0, 0);
    if (nk > 1) issueTile(1, 1);

    v8f acc[2][4] = {};

    int cb = 0;               // compute buffer = kt % NBUF
    for (int kt = 0; kt < nk; ++kt) {
        // own tile-kt chunks landed (tile kt+1 may stay in flight) ...
        if (kt + 1 < nk) waitAsync<4>(); else waitAsync<0>();
        // ... everyone's tile-kt chunks landed AND everyone done reading buf (kt-1)%NBUF
        __syncthreads();

        v16bf af0 = ldsFrag(lA[cb], waveM,      lane);
        v16bf af1 = ldsFrag(lA[cb], waveM + 16, lane);
        v16bf bf0 = ldsFrag(lB[cb], waveN,      lane);
        v16bf bf1 = ldsFrag(lB[cb], waveN + 16, lane);
        v16bf bf2 = ldsFrag(lB[cb], waveN + 32, lane);
        v16bf bf3 = ldsFrag(lB[cb], waveN + 48, lane);

        acc[0][0] = __builtin_amdgcn_wmma_f32_16x16x32_bf16(false, af0, false, bf0, (short)0, acc[0][0], false, false);
        acc[0][1] = __builtin_amdgcn_wmma_f32_16x16x32_bf16(false, af0, false, bf1, (short)0, acc[0][1], false, false);
        acc[0][2] = __builtin_amdgcn_wmma_f32_16x16x32_bf16(false, af0, false, bf2, (short)0, acc[0][2], false, false);
        acc[0][3] = __builtin_amdgcn_wmma_f32_16x16x32_bf16(false, af0, false, bf3, (short)0, acc[0][3], false, false);
        acc[1][0] = __builtin_amdgcn_wmma_f32_16x16x32_bf16(false, af1, false, bf0, (short)0, acc[1][0], false, false);
        acc[1][1] = __builtin_amdgcn_wmma_f32_16x16x32_bf16(false, af1, false, bf1, (short)0, acc[1][1], false, false);
        acc[1][2] = __builtin_amdgcn_wmma_f32_16x16x32_bf16(false, af1, false, bf2, (short)0, acc[1][2], false, false);
        acc[1][3] = __builtin_amdgcn_wmma_f32_16x16x32_bf16(false, af1, false, bf3, (short)0, acc[1][3], false, false);

        // refill the buffer freed by iteration kt-1 (safe: barrier above proves all reads done)
        if (kt + 2 < nk) {
            int ib = cb + 2; if (ib >= NBUF) ib -= NBUF;
            issueTile(kt + 2, ib);
        }
        ++cb; if (cb == NBUF) cb = 0;
    }

    // epilogue: C 16x16 layout — lane: n = L%16; VGPR r: m = (L/16)*8 + r
    #pragma unroll
    for (int i = 0; i < 2; ++i) {
        const int mBase = m0 + waveM + i * 16 + (lane >> 4) * 8;
        #pragma unroll
        for (int j = 0; j < 4; ++j) {
            const int n  = n0 + waveN + j * 16 + (lane & 15);
            const float bv = HAS_BIAS ? bptr[n] : 0.0f;
            #pragma unroll
            for (int r = 0; r < 8; ++r) {
                const float val = acc[i][j][r] + bv;
                const long long mg = mBase + r;
                if (OUT_BF16) {
                    if (TRANS_OUT) Cb[(long long)n * ldc + mg] = f2bf(val);
                    else           Cb[mg * (long long)ldc + n] = f2bf(val);
                } else {
                    Cf[mg * (long long)ldc + n] = val;
                }
            }
        }
    }
}

// Row softmax over bf16 scores in place, with pre-scale (1/D^2 per reference).
__global__ __launch_bounds__(256)
void softmax_rows(u16* __restrict__ P, int rowLen, float scale)
{
    __shared__ float red[8];
    u16* p = P + (long long)blockIdx.x * rowLen;
    const int tid  = threadIdx.x;
    const int lane = tid & 31;
    const int wave = tid >> 5;

    float x[4];
    #pragma unroll
    for (int i = 0; i < 4; ++i) x[i] = bf2f(p[tid + i * 256]) * scale;

    float m = fmaxf(fmaxf(x[0], x[1]), fmaxf(x[2], x[3]));
    #pragma unroll
    for (int off = 16; off; off >>= 1) m = fmaxf(m, __shfl_xor(m, off, 32));
    if (lane == 0) red[wave] = m;
    __syncthreads();
    m = red[0];
    #pragma unroll
    for (int w = 1; w < 8; ++w) m = fmaxf(m, red[w]);
    __syncthreads();

    float e[4], s = 0.0f;
    #pragma unroll
    for (int i = 0; i < 4; ++i) { e[i] = __expf(x[i] - m); s += e[i]; }
    #pragma unroll
    for (int off = 16; off; off >>= 1) s += __shfl_xor(s, off, 32);
    if (lane == 0) red[wave] = s;
    __syncthreads();
    s = 0.0f;
    #pragma unroll
    for (int w = 0; w < 8; ++w) s += red[w];

    const float inv = 1.0f / s;
    #pragma unroll
    for (int i = 0; i < 4; ++i) p[tid + i * 256] = f2bf(e[i] * inv);
}

__global__ void f32_to_bf16_kernel(const float* __restrict__ src, u16* __restrict__ dst, long long n4)
{
    long long i = (long long)blockIdx.x * blockDim.x + threadIdx.x;
    if (i >= n4) return;
    float4 v = reinterpret_cast<const float4*>(src)[i];
    uint2 o;
    o.x = (unsigned)f2bf(v.x) | ((unsigned)f2bf(v.y) << 16);
    o.y = (unsigned)f2bf(v.z) | ((unsigned)f2bf(v.w) << 16);
    reinterpret_cast<uint2*>(dst)[i] = o;
}

extern "C" void kernel_launch(void* const* d_in, const int* in_sizes, int n_in,
                              void* d_out, int out_size, void* d_ws, size_t ws_size,
                              hipStream_t stream)
{
    (void)in_sizes; (void)n_in; (void)out_size; (void)ws_size;
    constexpr int Bn = 8, S = 1024, D = 768, H = 12, HD = H * D;

    const float* X  = (const float*)d_in[0];
    const float* Wq = (const float*)d_in[1];
    const float* bq = (const float*)d_in[2];
    const float* Wk = (const float*)d_in[3];
    const float* bk = (const float*)d_in[4];
    const float* Wv = (const float*)d_in[5];
    const float* bv = (const float*)d_in[6];
    const float* Wo = (const float*)d_in[7];
    const float* bo = (const float*)d_in[8];

    char* wsp = (char*)d_ws;
    auto alloc = [&](size_t bytes) -> char* {
        char* p = wsp; wsp += (bytes + 255) & ~(size_t)255; return p;
    };
    const size_t nX  = (size_t)Bn * S * D;
    const size_t nW  = (size_t)H * D * D;
    const size_t nWo = (size_t)D * HD;
    const size_t nQ  = (size_t)Bn * H * S * D;
    const size_t nP  = (size_t)Bn * H * S * S;
    const size_t nHc = (size_t)Bn * S * HD;

    u16* Xb  = (u16*)alloc(nX  * 2);
    u16* Wqb = (u16*)alloc(nW  * 2);
    u16* Wkb = (u16*)alloc(nW  * 2);
    u16* Wvb = (u16*)alloc(nW  * 2);
    u16* Wob = (u16*)alloc(nWo * 2);
    u16* Q   = (u16*)alloc(nQ  * 2);
    u16* Kb  = (u16*)alloc(nQ  * 2);
    u16* VT  = (u16*)alloc(nQ  * 2);  // stored transposed: [B,H,D,S]
    u16* P   = (u16*)alloc(nP  * 2);
    u16* Hc  = (u16*)alloc(nHc * 2);  // [B,S,H*D]

    auto cvt = [&](const float* s, u16* d, size_t n) {
        long long n4 = (long long)(n / 4);
        f32_to_bf16_kernel<<<dim3((unsigned)((n4 + 255) / 256)), dim3(256), 0, stream>>>(s, d, n4);
    };
    cvt(X, Xb, nX); cvt(Wq, Wqb, nW); cvt(Wk, Wkb, nW); cvt(Wv, Wvb, nW); cvt(Wo, Wob, nWo);

    dim3 blk(256);

    // Q/K/V: per (b,h): [S,D] = Xb[b] (SxD) * W[h]^T (DxD) + bias[h]
    {
        dim3 grid(D / BN, S / BM, Bn * H);
        gemm_bf16_wmma<true, false, true><<<grid, blk, 0, stream>>>(Xb, Wqb, Q, bq,
            S, D, D, D, D, D,
            H, (long long)S * D, 0LL, 0LL, (long long)D * D,
            (long long)H * S * D, (long long)S * D, D);
        gemm_bf16_wmma<true, false, true><<<grid, blk, 0, stream>>>(Xb, Wkb, Kb, bk,
            S, D, D, D, D, D,
            H, (long long)S * D, 0LL, 0LL, (long long)D * D,
            (long long)H * S * D, (long long)S * D, D);
        gemm_bf16_wmma<true, true, true><<<grid, blk, 0, stream>>>(Xb, Wvb, VT, bv,
            S, D, D, D, D, /*ldc(transposed)=*/S,
            H, (long long)S * D, 0LL, 0LL, (long long)D * D,
            (long long)H * D * S, (long long)D * S, D);
    }

    // raw scores: P[b,h] = Q[b,h] (SxD) * K[b,h]^T
    {
        dim3 grid(S / BN, S / BM, Bn * H);
        gemm_bf16_wmma<true, false, false><<<grid, blk, 0, stream>>>(Q, Kb, P, nullptr,
            S, S, D, D, D, S,
            1, (long long)S * D, 0LL, (long long)S * D, 0LL,
            (long long)S * S, 0LL, 0);
    }

    // softmax over rows with reference's 1/D^2 scaling
    softmax_rows<<<dim3(Bn * H * S), dim3(256), 0, stream>>>(P, S, 1.0f / ((float)D * (float)D));

    // Hcat[b,:,h*D:(h+1)*D] = P[b,h] (SxS) * V[b,h] ; B operand = VT [D,S]
    {
        dim3 grid(D / BN, S / BM, Bn * H);
        gemm_bf16_wmma<true, false, false><<<grid, blk, 0, stream>>>(P, VT, Hc, nullptr,
            S, D, S, S, S, HD,
            H, (long long)H * S * S, (long long)S * S,
            (long long)H * D * S, (long long)D * S,
            (long long)S * HD, (long long)D, 0);
    }

    // out[b] = Hc[b] (S x HD) * Wo^T (HD x D) + bo  (f32 output)
    {
        dim3 grid(D / BN, S / BM, Bn);
        gemm_bf16_wmma<false, false, true><<<grid, blk, 0, stream>>>(Hc, Wob, d_out, bo,
            S, D, HD, HD, HD, D,
            1, (long long)S * HD, 0LL, 0LL, 0LL,
            (long long)S * D, 0LL, 0);
    }
}
